// MixtureOfExperts_90563680403698
// MI455X (gfx1250) — matrix operand order repaired and checked by reading, not compiled, not run on hardware
//
#include <hip/hip_runtime.h>

// ---------------------------------------------------------------------------
// MoE: out[b,c] = sum_e softmax(x@Wg+bg)[b,e] * (relu(x@W1[e]+b1[e]) @ W2[e] + b2[e])[c]
// B=65536, D=512, E=8, H=256, C=101.  f16 WMMA (16x16x32), f32 accumulate.
// ---------------------------------------------------------------------------

#define B_TOT 65536
#define D_DIM 512
#define C_DIM 101
#define E_DIM 8
#define H_DIM 256
#define N2    128          // C padded to 8 WMMA tiles
#define M_TILE 32          // rows per block
#define XSTR  520          // LDS stride for x tile (512 + 8 halves pad)
#define HSTR  264          // LDS stride for h tile (256 + 8 halves pad)

typedef __attribute__((ext_vector_type(16))) _Float16 v16h;
typedef __attribute__((ext_vector_type(8)))  _Float16 v8h;
typedef __attribute__((ext_vector_type(4)))  _Float16 v4h;
typedef __attribute__((ext_vector_type(8)))  float    v8f;

union V16 { v16h v; v8h h[2]; };

// ---------------------------------------------------------------------------
// Pack W1 [E,D,H] f32 -> B-fragment-layout f16.
// Fragment index id = ((e*16 + kb)*256 + n)*2 + p ; 16 halves each.
// B-matrix lane layout (ISA 7.12.2): lane part p covers K = kb*32 + 16*(h/8) + 8*p + h%8.
// ---------------------------------------------------------------------------
__global__ __launch_bounds__(256) void pack_w1(const float* __restrict__ W1,
                                               _Float16* __restrict__ w1p) {
    int id = blockIdx.x * 256 + threadIdx.x;     // 65536 fragments
    int p  = id & 1;
    int n  = (id >> 1) & 255;
    int kb = (id >> 9) & 15;
    int e  = id >> 13;
    v16h frag;
#pragma unroll
    for (int h = 0; h < 16; ++h) {
        int k = kb * 32 + (h & 8) * 2 + 8 * p + (h & 7);
        frag[h] = (_Float16)W1[((size_t)(e * D_DIM) + k) * H_DIM + n];
    }
    *(v16h*)(w1p + (size_t)id * 16) = frag;
}

// Pack W2 [E,H,C] f32 -> padded [E, kb=8, n=128, p=2] f16 fragments (zeros for n>=101)
__global__ __launch_bounds__(256) void pack_w2(const float* __restrict__ W2,
                                               _Float16* __restrict__ w2p) {
    int id = blockIdx.x * 256 + threadIdx.x;     // 16384 fragments
    int p  = id & 1;
    int n  = (id >> 1) & 127;
    int kb = (id >> 8) & 7;
    int e  = id >> 11;
    v16h frag;
#pragma unroll
    for (int h = 0; h < 16; ++h) {
        int k = kb * 32 + (h & 8) * 2 + 8 * p + (h & 7);
        float v = (n < C_DIM) ? W2[((size_t)(e * H_DIM) + k) * C_DIM + n] : 0.0f;
        frag[h] = (_Float16)v;
    }
    *(v16h*)(w2p + (size_t)id * 16) = frag;
}

// ---------------------------------------------------------------------------
// Gate: gates[b,:] = softmax(x[b,:] @ Wg + bg).  One wave per row (wave32).
// ---------------------------------------------------------------------------
__global__ __launch_bounds__(256) void gate_kernel(const float* __restrict__ x,
                                                   const float* __restrict__ Wg,
                                                   const float* __restrict__ bg,
                                                   float* __restrict__ gates) {
    __shared__ float wgl[D_DIM * E_DIM];   // 16 KB
    __shared__ float bgl[E_DIM];
    int tid = threadIdx.x;
    for (int i = tid; i < D_DIM * E_DIM; i += 256) wgl[i] = Wg[i];
    if (tid < E_DIM) bgl[tid] = bg[tid];
    __syncthreads();

    int wave = tid >> 5;
    int lane = tid & 31;
    size_t row = (size_t)blockIdx.x * 8 + wave;
    const float* xr = x + row * D_DIM;

    float acc[E_DIM] = {0.f, 0.f, 0.f, 0.f, 0.f, 0.f, 0.f, 0.f};
    for (int k = lane; k < D_DIM; k += 32) {
        float xv = xr[k];
        const float* w = wgl + k * E_DIM;
#pragma unroll
        for (int e = 0; e < E_DIM; ++e) acc[e] = fmaf(xv, w[e], acc[e]);
    }
#pragma unroll
    for (int e = 0; e < E_DIM; ++e)
        for (int off = 16; off > 0; off >>= 1)
            acc[e] += __shfl_down(acc[e], off, 32);

    if (lane == 0) {
        float z[E_DIM];
        float m = -3.0e38f;
#pragma unroll
        for (int e = 0; e < E_DIM; ++e) { z[e] = acc[e] + bgl[e]; m = fmaxf(m, z[e]); }
        float s = 0.f;
#pragma unroll
        for (int e = 0; e < E_DIM; ++e) { z[e] = __expf(z[e] - m); s += z[e]; }
        float inv = 1.f / s;
#pragma unroll
        for (int e = 0; e < E_DIM; ++e) gates[row * E_DIM + e] = z[e] * inv;
    }
}

// ---------------------------------------------------------------------------
// Main fused MoE kernel. 256 threads = 8 waves; block tile = 32 rows.
// Wave nj (0..7) owns an N-strip: 32 cols of H (layer 1) / 16 cols of padded C
// (layer 2), and computes BOTH 16-row M-strips, so every B fragment feeds two
// WMMAs and weight traffic from L2 is halved.
// ---------------------------------------------------------------------------
__global__ __launch_bounds__(256) void moe_main(const float* __restrict__ x,
                                                const float* __restrict__ b1,
                                                const float* __restrict__ b2,
                                                const float* __restrict__ gates,
                                                const _Float16* __restrict__ w1p,
                                                const _Float16* __restrict__ w2p,
                                                float* __restrict__ out) {
    __shared__ _Float16 xs[M_TILE * XSTR];   // 33280 B
    __shared__ _Float16 hs[M_TILE * HSTR];   // 16896 B
    __shared__ float    gs[M_TILE * E_DIM];  //  1024 B

    const int tid  = threadIdx.x;
    const int row0 = blockIdx.x * M_TILE;

    // ---- stage x tile (f32 -> f16) into LDS, float4 loads, b64 DS stores ----
#pragma unroll
    for (int it = 0; it < (M_TILE * D_DIM / 4) / 256; ++it) {
        int i4 = tid + it * 256;             // over 4096 float4
        int r  = i4 >> 7;
        int c4 = i4 & 127;
        float4 v = ((const float4*)(x + (size_t)(row0 + r) * D_DIM))[c4];
        v4h hv;
        hv[0] = (_Float16)v.x; hv[1] = (_Float16)v.y;
        hv[2] = (_Float16)v.z; hv[3] = (_Float16)v.w;
        *(v4h*)(xs + r * XSTR + c4 * 4) = hv;
    }
    gs[tid] = gates[(size_t)row0 * E_DIM + tid];   // 256 == M_TILE*E_DIM
    __syncthreads();

    const int nj   = tid >> 5;        // wave id = N-strip 0..7
    const int lane = tid & 31;
    const int ln   = lane & 15;       // column / row-in-strip selector
    const int lp   = lane >> 4;       // lane part (K split)

    // hoisted per-lane pointers (constant offsets inside the loops)
    const _Float16* arow0 = xs + ln * XSTR + 8 * lp;              // A strip 0
    const _Float16* arow1 = xs + (16 + ln) * XSTR + 8 * lp;       // A strip 1
    const _Float16* hrow0 = hs + ln * HSTR + 8 * lp;
    const _Float16* hrow1 = hs + (16 + ln) * HSTR + 8 * lp;
    _Float16*       hst0  = hs + (8 * lp) * HSTR + nj * 32 + ln;
    _Float16*       hst1  = hs + (16 + 8 * lp) * HSTR + nj * 32 + ln;
    const float*    gsl0  = gs + (8 * lp) * E_DIM;
    const float*    gsl1  = gs + (16 + 8 * lp) * E_DIM;
    float*          outp0 = out + (size_t)(row0 + 8 * lp) * C_DIM + nj * 16 + ln;
    float*          outp1 = out + (size_t)(row0 + 16 + 8 * lp) * C_DIM + nj * 16 + ln;
    const v16h*     w1lane = (const v16h*)w1p + ((nj * 32 + ln) * 2 + lp);
    const v16h*     w2lane = (const v16h*)w2p + ((nj * 16 + ln) * 2 + lp);

    const v8f zerov = {0.f, 0.f, 0.f, 0.f, 0.f, 0.f, 0.f, 0.f};
    v8f outacc[2];                    // [m-strip]
    outacc[0] = zerov; outacc[1] = zerov;

    for (int e = 0; e < E_DIM; ++e) {
        const v16h* w1e = w1lane + (size_t)e * (16 * 256 * 2);
        const v16h* w2e = w2lane + (size_t)e * (8 * 128 * 2);

        // ---------------- layer 1: h = relu(x @ W1[e] + b1[e]) ----------------
        v8f acc1[2][2];               // [m-strip][n-tile]
#pragma unroll
        for (int s = 0; s < 2; ++s) { acc1[s][0] = zerov; acc1[s][1] = zerov; }

#pragma unroll
        for (int kb = 0; kb < 16; ++kb) {
            V16 a0, a1;
            a0.h[0] = *(const v8h*)(arow0 + kb * 32);
            a0.h[1] = *(const v8h*)(arow0 + kb * 32 + 16);
            a1.h[0] = *(const v8h*)(arow1 + kb * 32);
            a1.h[1] = *(const v8h*)(arow1 + kb * 32 + 16);
            v16h b0 = w1e[kb * 512];
            v16h b1f = w1e[kb * 512 + 32];
            acc1[0][0] = __builtin_amdgcn_wmma_f32_16x16x32_f16(
                false, a0.v, false, b0, (short)0, acc1[0][0], false, false);
            acc1[1][0] = __builtin_amdgcn_wmma_f32_16x16x32_f16(
                false, a1.v, false, b0, (short)0, acc1[1][0], false, false);
            acc1[0][1] = __builtin_amdgcn_wmma_f32_16x16x32_f16(
                false, a0.v, false, b1f, (short)0, acc1[0][1], false, false);
            acc1[1][1] = __builtin_amdgcn_wmma_f32_16x16x32_f16(
                false, a1.v, false, b1f, (short)0, acc1[1][1], false, false);
        }

        __syncthreads();   // previous expert's layer-2 reads of hs are done
        // bias + relu, write h strips to LDS as f16 (C/D layout: m = v + 8*lp)
#pragma unroll
        for (int t = 0; t < 2; ++t) {
            float b1v = b1[e * H_DIM + nj * 32 + t * 16 + ln];
#pragma unroll
            for (int v = 0; v < 8; ++v) {
                float h0 = fmaxf(acc1[0][t][v] + b1v, 0.f);
                float h1 = fmaxf(acc1[1][t][v] + b1v, 0.f);
                hst0[v * HSTR + t * 16] = (_Float16)h0;
                hst1[v * HSTR + t * 16] = (_Float16)h1;
            }
        }
        __syncthreads();   // hs visible to all waves

        // ---------------- layer 2: eo = h @ W2[e]  (N padded to 128) ----------
        v8f acc2[2];
        acc2[0] = zerov; acc2[1] = zerov;
#pragma unroll
        for (int kb = 0; kb < 8; ++kb) {
            V16 a0, a1;
            a0.h[0] = *(const v8h*)(hrow0 + kb * 32);
            a0.h[1] = *(const v8h*)(hrow0 + kb * 32 + 16);
            a1.h[0] = *(const v8h*)(hrow1 + kb * 32);
            a1.h[1] = *(const v8h*)(hrow1 + kb * 32 + 16);
            v16h bf = w2e[kb * 256];
            acc2[0] = __builtin_amdgcn_wmma_f32_16x16x32_f16(
                false, a0.v, false, bf, (short)0, acc2[0], false, false);
            acc2[1] = __builtin_amdgcn_wmma_f32_16x16x32_f16(
                false, a1.v, false, bf, (short)0, acc2[1], false, false);
        }

        // gate-weighted accumulation into running output
        int n = nj * 16 + ln;
        float b2v = (n < C_DIM) ? b2[e * C_DIM + n] : 0.f;
#pragma unroll
        for (int v = 0; v < 8; ++v) {
            float g0 = gsl0[v * E_DIM + e];
            float g1 = gsl1[v * E_DIM + e];
            outacc[0][v] += g0 * (acc2[0][v] + b2v);
            outacc[1][v] += g1 * (acc2[1][v] + b2v);
        }
    }

    // ---- store out[row, c] for c < 101 ----
    int n = nj * 16 + ln;
    if (n < C_DIM) {
#pragma unroll
        for (int v = 0; v < 8; ++v) {
            outp0[v * C_DIM] = outacc[0][v];
            outp1[v * C_DIM] = outacc[1][v];
        }
    }
}

// ---------------------------------------------------------------------------
extern "C" void kernel_launch(void* const* d_in, const int* in_sizes, int n_in,
                              void* d_out, int out_size, void* d_ws, size_t ws_size,
                              hipStream_t stream) {
    (void)in_sizes; (void)n_in; (void)out_size; (void)ws_size;
    const float* x  = (const float*)d_in[0];
    const float* W1 = (const float*)d_in[1];
    const float* b1 = (const float*)d_in[2];
    const float* W2 = (const float*)d_in[3];
    const float* b2 = (const float*)d_in[4];
    const float* Wg = (const float*)d_in[5];
    const float* bg = (const float*)d_in[6];
    float* out = (float*)d_out;

    char* ws = (char*)d_ws;
    _Float16* w1p  = (_Float16*)ws;                              // 4 MB
    _Float16* w2p  = (_Float16*)(ws + (4u << 20));               // 512 KB
    float*    gts  = (float*)(ws + (4u << 20) + (512u << 10));   // 2 MB

    pack_w1<<<E_DIM * 16 * 256 * 2 / 256, 256, 0, stream>>>(W1, w1p);
    pack_w2<<<E_DIM * 8 * 128 * 2 / 256, 256, 0, stream>>>(W2, w2p);
    gate_kernel<<<B_TOT / 8, 256, 0, stream>>>(x, Wg, bg, gts);
    moe_main<<<B_TOT / M_TILE, 256, 0, stream>>>(x, b1, b2, gts, w1p, w2p, out);
}
